// FP8LinearWrapper_33844342292642
// MI455X (gfx1250) — compile-verified, matrix operand order checked by reading, not run on hardware
//
#include <hip/hip_runtime.h>

// ---------------------------------------------------------------------------
// Types
// ---------------------------------------------------------------------------
typedef __attribute__((ext_vector_type(16))) int   v16i;
typedef __attribute__((ext_vector_type(8)))  float v8f;

// ---------------------------------------------------------------------------
// Kernel 1: zero the atomic-max slots (must run every call; harness poisons ws)
// ---------------------------------------------------------------------------
__global__ void fp8lin_init_kernel(unsigned int* slots) {
    slots[0] = 0u;
    slots[1] = 0u;
}

// ---------------------------------------------------------------------------
// Kernel 2: grid-stride abs-max reduction (float4 loads, LDS reduce, atomicMax
// on uint bit pattern — valid ordering for non-negative floats)
// ---------------------------------------------------------------------------
__global__ __launch_bounds__(256)
void fp8lin_absmax_kernel(const float* __restrict__ p, size_t n4,
                          unsigned int* __restrict__ slot) {
    size_t i      = (size_t)blockIdx.x * blockDim.x + threadIdx.x;
    size_t stride = (size_t)gridDim.x * blockDim.x;
    float m = 0.0f;
    const float4* p4 = (const float4*)p;
    for (size_t k = i; k < n4; k += stride) {
        float4 v = p4[k];
        m = fmaxf(m, fmaxf(fmaxf(fabsf(v.x), fabsf(v.y)),
                           fmaxf(fabsf(v.z), fabsf(v.w))));
    }
    __shared__ float red[256];
    red[threadIdx.x] = m;
    __syncthreads();
    for (int s = 128; s > 0; s >>= 1) {
        if ((int)threadIdx.x < s)
            red[threadIdx.x] = fmaxf(red[threadIdx.x], red[threadIdx.x + s]);
        __syncthreads();
    }
    if (threadIdx.x == 0)
        atomicMax(slot, __float_as_uint(red[0]));
}

// ---------------------------------------------------------------------------
// Kernel 3: compute scales.
//   scales[0] = 1/x_scale  (448/x_max, or 1 if x_max==0)
//   scales[1] = 1/w_scale  (448/w_max)
//   scales[2] = x_scale * w_scale  (dequant factor for the GEMM epilogue)
// ---------------------------------------------------------------------------
__global__ void fp8lin_scale_kernel(const unsigned int* __restrict__ slots,
                                    float* __restrict__ scales) {
    const float FP8_MAX = 448.0f;
    float xmax = __uint_as_float(slots[0]);
    float wmax = __uint_as_float(slots[1]);
    float xs = (xmax > 0.0f) ? (xmax / FP8_MAX) : 1.0f;
    float ws = wmax / FP8_MAX;
    scales[0] = 1.0f / xs;
    scales[1] = (ws != 0.0f) ? (1.0f / ws) : 0.0f;
    scales[2] = xs * ws;
}

// ---------------------------------------------------------------------------
// float -> fp8 e4m3 (OCP fn) packing: prefer hardware v_cvt_pk_fp8_f32.
// ---------------------------------------------------------------------------
#if __has_builtin(__builtin_amdgcn_cvt_pk_fp8_f32)
#define HAVE_HW_FP8_CVT 1
#else
#define HAVE_HW_FP8_CVT 0
#endif

#if !HAVE_HW_FP8_CVT
__device__ inline unsigned int f32_to_e4m3_sw(float f) {
    unsigned int u    = __float_as_uint(f);
    unsigned int sign = (u >> 24) & 0x80u;
    unsigned int expf = (u >> 23) & 0xFFu;
    unsigned int man  = u & 0x7FFFFFu;
    if (expf == 0xFFu) return sign | 0x7Eu;            // inf/nan -> +-448
    int e = (int)expf - 127;
    if (e > 8) return sign | 0x7Eu;                    // overflow -> max
    if (e < -10) return sign;                          // underflow -> 0
    unsigned int full = man | 0x800000u;
    int shift, ebits;
    if (e < -6) { shift = 20 + (-6 - e); ebits = 0; }  // subnormal target
    else        { shift = 20;            ebits = e + 7; }
    unsigned int lsb    = full >> shift;
    unsigned int rbit   = (full >> (shift - 1)) & 1u;
    unsigned int sticky = (full & ((1u << (shift - 1)) - 1u)) ? 1u : 0u;
    unsigned int mant   = lsb + (rbit & (sticky | (lsb & 1u)));   // RNE
    unsigned int code   = ((unsigned int)ebits << 3) + mant;      // carry ok
    if (code > 0x7Eu) code = 0x7Eu;                    // clamp to 448
    return sign | code;
}
#endif

__device__ inline unsigned int pack4_e4m3(float a, float b, float c, float d) {
#if HAVE_HW_FP8_CVT
    int p = __builtin_amdgcn_cvt_pk_fp8_f32(a, b, 0, false);
    p     = __builtin_amdgcn_cvt_pk_fp8_f32(c, d, p, true);
    return (unsigned int)p;
#else
    return f32_to_e4m3_sw(a) | (f32_to_e4m3_sw(b) << 8) |
           (f32_to_e4m3_sw(c) << 16) | (f32_to_e4m3_sw(d) << 24);
#endif
}

// ---------------------------------------------------------------------------
// Kernel 4: quantize fp32 -> fp8 (16 elements / thread, b128 store)
// ---------------------------------------------------------------------------
__global__ __launch_bounds__(256)
void fp8lin_quant_kernel(const float* __restrict__ src,
                         unsigned char* __restrict__ dst,
                         size_t n16, const float* __restrict__ scales,
                         int which) {
    const float inv = scales[which];
    size_t i      = (size_t)blockIdx.x * blockDim.x + threadIdx.x;
    size_t stride = (size_t)gridDim.x * blockDim.x;
    const float4* s4 = (const float4*)src;
    uint4*        d4 = (uint4*)dst;
    for (size_t k = i; k < n16; k += stride) {
        float4 a = s4[4 * k + 0];
        float4 b = s4[4 * k + 1];
        float4 c = s4[4 * k + 2];
        float4 d = s4[4 * k + 3];
        uint4 o;
        o.x = pack4_e4m3(a.x * inv, a.y * inv, a.z * inv, a.w * inv);
        o.y = pack4_e4m3(b.x * inv, b.y * inv, b.z * inv, b.w * inv);
        o.z = pack4_e4m3(c.x * inv, c.y * inv, c.z * inv, c.w * inv);
        o.w = pack4_e4m3(d.x * inv, d.y * inv, d.z * inv, d.w * inv);
        d4[k] = o;   // fp8 operands stay cache-resident (re-read by GEMM)
    }
}

// ---------------------------------------------------------------------------
// Kernel 5: FP8 GEMM with V_WMMA_F32_16X16X128_FP8_FP8.
// One wave computes a 64(M) x 64(N) output tile:
//   4 A fragments (16x128, streamed) x 4 B fragments (128x16, resident)
//   -> 16 f32 accumulators, 16 WMMAs per 128-deep K step (~64 FLOP/byte).
// The 8 waves of a block tile a 128(M) x 256(N) super-tile (2x4 wave grid)
// so B fragments are shared through WGP$/L2.
// x is [M,K] row-major, W is [N,K] row-major -> both fragments are 64
// contiguous bytes per lane per K-step (natural v16i loads, no transpose).
// ---------------------------------------------------------------------------
__global__ __launch_bounds__(256)
void fp8lin_gemm_kernel(const unsigned char* __restrict__ xq,
                        const unsigned char* __restrict__ wq,
                        const float* __restrict__ bias,
                        const float* __restrict__ scale2,
                        float* __restrict__ out,
                        int M, int N, int K) {
    const int lane = threadIdx.x & 31;
    const int wave = threadIdx.x >> 5;

    const int tilesM = (M + 63) >> 6;          // 64-row M tiles
    const int superM = (tilesM + 1) >> 1;      // block spans 2 M tiles
    const int sm = blockIdx.x % superM;
    const int sn = blockIdx.x / superM;

    const int tm = sm * 2 + (wave >> 2);       // 64-row M tile index
    const int tn = sn * 4 + (wave & 3);        // 64-col N tile index
    if (tm * 64 >= M) return;                  // wave-uniform: EXEC all-ones

    const int r = lane & 15;                   // row within fragment
    const int h = lane >> 4;                   // K-half selector (0 or 1)

    const unsigned char* pa[4];
    const unsigned char* pb[4];
    #pragma unroll
    for (int i = 0; i < 4; ++i) {
        pa[i] = xq + (size_t)(tm * 64 + i * 16 + r) * K + h * 64;
        pb[i] = wq + (size_t)(tn * 64 + i * 16 + r) * K + h * 64;
    }

    v8f acc[4][4] = {};

    for (int k = 0; k < K; k += 128) {
        if (k + 128 < K) {                     // prime near cache for next slab
            __builtin_prefetch(pa[0] + k + 128, 0, 3);
            __builtin_prefetch(pa[2] + k + 128, 0, 3);
            __builtin_prefetch(pb[0] + k + 128, 0, 3);
            __builtin_prefetch(pb[2] + k + 128, 0, 3);
        }
        const v16i b0 = *(const v16i*)(pb[0] + k);
        const v16i b1 = *(const v16i*)(pb[1] + k);
        const v16i b2 = *(const v16i*)(pb[2] + k);
        const v16i b3 = *(const v16i*)(pb[3] + k);
        #pragma unroll
        for (int i = 0; i < 4; ++i) {
            const v16i a = *(const v16i*)(pa[i] + k);
            acc[i][0] = __builtin_amdgcn_wmma_f32_16x16x128_fp8_fp8(
                a, b0, (short)0, acc[i][0], false, false);
            acc[i][1] = __builtin_amdgcn_wmma_f32_16x16x128_fp8_fp8(
                a, b1, (short)0, acc[i][1], false, false);
            acc[i][2] = __builtin_amdgcn_wmma_f32_16x16x128_fp8_fp8(
                a, b2, (short)0, acc[i][2], false, false);
            acc[i][3] = __builtin_amdgcn_wmma_f32_16x16x128_fp8_fp8(
                a, b3, (short)0, acc[i][3], false, false);
        }
    }

    // Epilogue: D layout — VGPR rr holds M = 8*(lane>>4) + rr, N = lane&15.
    // out is write-once streaming data: use non-temporal stores so the fp8
    // operands (L2-resident, re-read by every tile) are not evicted.
    const float cs = *scale2;
    const int   col_in = lane & 15;
    #pragma unroll
    for (int i = 0; i < 4; ++i) {
        #pragma unroll
        for (int j = 0; j < 4; ++j) {
            const int colg = tn * 64 + j * 16 + col_in;
            const float bv = bias[colg];
            float* po = out + (size_t)(tm * 64 + i * 16 + h * 8) * N + colg;
            #pragma unroll
            for (int rr = 0; rr < 8; ++rr) {
                __builtin_nontemporal_store(acc[i][j][rr] * cs + bv,
                                            po + (size_t)rr * N);
            }
        }
    }
}

// ---------------------------------------------------------------------------
// Host launcher
// ---------------------------------------------------------------------------
extern "C" void kernel_launch(void* const* d_in, const int* in_sizes, int n_in,
                              void* d_out, int out_size, void* d_ws, size_t ws_size,
                              hipStream_t stream) {
    const float* x    = (const float*)d_in[0];
    const float* W    = (const float*)d_in[1];
    const float* bias = (const float*)d_in[2];
    float*       out  = (float*)d_out;

    const int Nout = in_sizes[2];                  // d_out  (4096)
    const int K    = in_sizes[1] / Nout;           // d_in   (4096)
    const int M    = in_sizes[0] / K;              // tokens (8192)

    // workspace layout
    unsigned int* slots  = (unsigned int*)d_ws;                    // 2 uints
    float*        scales = (float*)((char*)d_ws + 64);             // 3 floats
    unsigned char* xq    = (unsigned char*)d_ws + 256;             // M*K bytes
    unsigned char* wq    = xq + (size_t)M * K;                     // N*K bytes

    fp8lin_init_kernel<<<1, 1, 0, stream>>>(slots);

    fp8lin_absmax_kernel<<<1024, 256, 0, stream>>>(x, ((size_t)M * K) >> 2, slots + 0);
    fp8lin_absmax_kernel<<<1024, 256, 0, stream>>>(W, ((size_t)Nout * K) >> 2, slots + 1);

    fp8lin_scale_kernel<<<1, 1, 0, stream>>>(slots, scales);

    fp8lin_quant_kernel<<<2048, 256, 0, stream>>>(x, xq, ((size_t)M * K) >> 4, scales, 0);
    fp8lin_quant_kernel<<<1024, 256, 0, stream>>>(W, wq, ((size_t)Nout * K) >> 4, scales, 1);

    const int tilesM = (M + 63) / 64;
    const int tilesN = Nout / 64;
    const int superM = (tilesM + 1) / 2;           // 2 M-tiles per block
    const int superN = (tilesN + 3) / 4;           // 4 N-tiles per block
    const int blocks = superM * superN;            // 8 waves (256 thr) / block
    fp8lin_gemm_kernel<<<blocks, 256, 0, stream>>>(xq, wq, bias, scales + 2,
                                                   out, M, Nout, K);
}
